// MovieRecommendationModel_withMovieEmbedding_6854767805142
// MI455X (gfx1250) — matrix operand order repaired and checked by reading, not compiled, not run on hardware
//
#include <hip/hip_runtime.h>
#include <hip/hip_bf16.h>

typedef __bf16 bf16_t;
typedef __attribute__((ext_vector_type(16))) __bf16 v16bf;
typedef __attribute__((ext_vector_type(8)))  __bf16 v8bf;
typedef __attribute__((ext_vector_type(8)))  float  v8f;

union V16 { v16bf v; v8bf h[2]; };

// ---------------- degree / normalization ----------------
__global__ void k_init_deg(float* deg, int n) {
    int i = blockIdx.x * blockDim.x + threadIdx.x;
    if (i < n) deg[i] = 1.0f;  // self-loop contributes 1 to every node's degree
}

__global__ void k_count_deg(const int* __restrict__ dst, float* deg, int E) {
    int e = blockIdx.x * blockDim.x + threadIdx.x;
    if (e < E) atomicAdd(&deg[dst[e]], 1.0f);
}

__global__ void k_finalize_dinv(float* deg, int n) {
    int i = blockIdx.x * blockDim.x + threadIdx.x;
    if (i < n) deg[i] = rsqrtf(deg[i]);
}

// ---------------- build initial features (bf16) ----------------
__global__ void k_build_h0(const int* __restrict__ x, const float* __restrict__ emb,
                           bf16_t* __restrict__ h0, int num_users, int d, int total) {
    int i = blockIdx.x * blockDim.x + threadIdx.x;
    if (i >= total) return;
    int row = i / d;
    int c   = i - row * d;
    int er  = (row < num_users) ? 0 : (x[row] - num_users + 1);
    h0[i] = (bf16_t)emb[(size_t)er * d + c];
}

// ---------------- weight f32(KxN) -> bf16 transposed (NxK) ----------------
__global__ void k_transpose_w(const float* __restrict__ W, bf16_t* __restrict__ Wt,
                              int K, int N) {
    int i = blockIdx.x * blockDim.x + threadIdx.x;
    if (i >= K * N) return;
    int k = i / N;
    int n = i - k * N;
    Wt[(size_t)n * K + k] = (bf16_t)W[i];
}

// ---------------- WMMA GEMM: C[M,N] = A[M,K] * Bt[N,K]^T (bf16 in, f32 out) ----
// One wave computes one 16x16 C tile; K multiple of 32.
// A-tile (from K-major A): lane l15 = row, 16 contiguous K starting at half*16.
// B-tile (from N-major Bt): lane l15 = column, 16 contiguous K starting at half*16.
__global__ void gemm_bf16_wmma(const bf16_t* __restrict__ A,
                               const bf16_t* __restrict__ Bt,
                               float* __restrict__ C,
                               int tilesM, int tilesN, int K, int ldc) {
    int gwave = (blockIdx.x * blockDim.x + threadIdx.x) >> 5;
    int lane  = threadIdx.x & 31;
    int tM = gwave / tilesN;
    int tN = gwave - tM * tilesN;
    if (tM >= tilesM) return;          // wave-uniform: EXEC stays all-1s for WMMA
    int half = lane >> 4;
    int l15  = lane & 15;
    const bf16_t* arow = A  + (size_t)(tM * 16 + l15) * K;
    const bf16_t* brow = Bt + (size_t)(tN * 16 + l15) * K;
    v8f acc = {};
    for (int k = 0; k < K; k += 32) {
        V16 ua, ub;
        ua.v    = *(const v16bf*)(arow + k + half * 16);
        ub.v    = *(const v16bf*)(brow + k + half * 16);
        acc = __builtin_amdgcn_wmma_f32_16x16x32_bf16(
            false, ua.v, false, ub.v, (short)0, acc, false, false);
    }
    float* cp = C + (size_t)(tM * 16 + half * 8) * ldc + tN * 16 + l15;
#pragma unroll
    for (int r = 0; r < 8; ++r) cp[(size_t)r * ldc] = acc[r];
}

// Final GEMM: C = clip(U @ I^T, 1, 5).  One wave computes a 16x64 strip (4 N-tiles),
// reusing the A fragment across 4 WMMAs per k-step.  Output is 268 MB streamed once:
// store non-temporal so it doesn't evict the hot U/I panels from L2.
__global__ void gemm_bf16_wmma_clip4(const bf16_t* __restrict__ A,
                                     const bf16_t* __restrict__ Bt,
                                     float* __restrict__ C,
                                     int tilesM, int tilesN4, int K, int ldc) {
    int gwave = (blockIdx.x * blockDim.x + threadIdx.x) >> 5;
    int lane  = threadIdx.x & 31;
    int tM  = gwave / tilesN4;
    int tN4 = gwave - tM * tilesN4;
    if (tM >= tilesM) return;          // wave-uniform exit
    int half = lane >> 4;
    int l15  = lane & 15;
    const bf16_t* arow = A + (size_t)(tM * 16 + l15) * K;
    const bf16_t* brow[4];
#pragma unroll
    for (int j = 0; j < 4; ++j)
        brow[j] = Bt + (size_t)(tN4 * 64 + j * 16 + l15) * K;

    v8f acc0 = {}, acc1 = {}, acc2 = {}, acc3 = {};
    for (int k = 0; k < K; k += 32) {
        V16 ua;
        ua.v = *(const v16bf*)(arow + k + half * 16);
        v16bf b0 = *(const v16bf*)(brow[0] + k + half * 16);
        v16bf b1 = *(const v16bf*)(brow[1] + k + half * 16);
        v16bf b2 = *(const v16bf*)(brow[2] + k + half * 16);
        v16bf b3 = *(const v16bf*)(brow[3] + k + half * 16);
        acc0 = __builtin_amdgcn_wmma_f32_16x16x32_bf16(false, ua.v, false, b0, (short)0, acc0, false, false);
        acc1 = __builtin_amdgcn_wmma_f32_16x16x32_bf16(false, ua.v, false, b1, (short)0, acc1, false, false);
        acc2 = __builtin_amdgcn_wmma_f32_16x16x32_bf16(false, ua.v, false, b2, (short)0, acc2, false, false);
        acc3 = __builtin_amdgcn_wmma_f32_16x16x32_bf16(false, ua.v, false, b3, (short)0, acc3, false, false);
    }
    float* cp = C + (size_t)(tM * 16 + half * 8) * ldc + tN4 * 64 + l15;
#pragma unroll
    for (int r = 0; r < 8; ++r) {
        float* row = cp + (size_t)r * ldc;
        __builtin_nontemporal_store(fminf(fmaxf(acc0[r], 1.0f), 5.0f), row);
        __builtin_nontemporal_store(fminf(fmaxf(acc1[r], 1.0f), 5.0f), row + 16);
        __builtin_nontemporal_store(fminf(fmaxf(acc2[r], 1.0f), 5.0f), row + 32);
        __builtin_nontemporal_store(fminf(fmaxf(acc3[r], 1.0f), 5.0f), row + 48);
    }
}

// ---------------- GCN aggregation ----------------
// self-loop term: agg = hw * dinv[row]^2   (also initializes agg -> no pre-zero needed)
__global__ void k_agg_self(const float* __restrict__ hw, const float* __restrict__ dinv,
                           float* __restrict__ agg, int Nf, int total) {
    int i = blockIdx.x * blockDim.x + threadIdx.x;
    if (i >= total) return;
    int row = i / Nf;
    float dv = dinv[row];
    agg[i] = hw[i] * dv * dv;
}

// edge scatter: agg[dst] += hw[src] * dinv[src]*dinv[dst]; one thread = (edge, 4 cols)
__global__ void k_agg_edges(const int* __restrict__ src, const int* __restrict__ dst,
                            const float* __restrict__ hw, const float* __restrict__ dinv,
                            float* __restrict__ agg, int E, int groups, int Nf) {
    long long idx = (long long)blockIdx.x * blockDim.x + threadIdx.x;
    int e = (int)(idx / groups);
    if (e >= E) return;
    int g = (int)(idx - (long long)e * groups);
    int s = src[e];
    int d = dst[e];
    float coef = dinv[s] * dinv[d];
    const float* hp = hw  + (size_t)s * Nf + g * 4;
    float*       ap = agg + (size_t)d * Nf + g * 4;
    atomicAdd(ap + 0, hp[0] * coef);
    atomicAdd(ap + 1, hp[1] * coef);
    atomicAdd(ap + 2, hp[2] * coef);
    atomicAdd(ap + 3, hp[3] * coef);
}

// out_bf16 = bf16(relu(agg + bias))   (Nf is a power of two)
__global__ void k_bias_relu_bf16(const float* __restrict__ agg, const float* __restrict__ b,
                                 bf16_t* __restrict__ out, int NfMask, int total) {
    int i = blockIdx.x * blockDim.x + threadIdx.x;
    if (i >= total) return;
    float v = agg[i] + b[i & NfMask];
    out[i] = (bf16_t)fmaxf(v, 0.0f);
}

extern "C" void kernel_launch(void* const* d_in, const int* in_sizes, int n_in,
                              void* d_out, int out_size, void* d_ws, size_t ws_size,
                              hipStream_t stream) {
    const int*   x   = (const int*)d_in[0];
    const int*   ei  = (const int*)d_in[1];   // (2,E): row0=src, row1=dst
    const float* emb = (const float*)d_in[3];
    const float* W1  = (const float*)d_in[4];
    const float* b1  = (const float*)d_in[5];
    const float* W2  = (const float*)d_in[6];
    const float* b2  = (const float*)d_in[7];
    const float* Wl  = (const float*)d_in[8];
    const float* bl  = (const float*)d_in[9];
    float* out = (float*)d_out;

    const int N  = in_sizes[0];
    const int E  = in_sizes[1] / 2;
    const int h1 = in_sizes[5];
    const int h2 = in_sizes[7];
    const int h3 = in_sizes[9];
    const int d  = in_sizes[4] / h1;
    const int num_items = in_sizes[3] / d - 1;
    const int num_users = N - num_items;
    const int maxF = (h1 > d) ? h1 : d;

    // workspace layout
    char* ws = (char*)d_ws;
    size_t off = 0;
    auto take = [&](size_t bytes) -> char* {
        char* p = ws + off;
        off = (off + bytes + 255) & ~(size_t)255;
        return p;
    };
    float*  dinv = (float*)take((size_t)N * 4);
    bf16_t* W1t  = (bf16_t*)take((size_t)d  * h1 * 2);
    bf16_t* W2t  = (bf16_t*)take((size_t)h1 * h2 * 2);
    bf16_t* Wlt  = (bf16_t*)take((size_t)h2 * h3 * 2);
    bf16_t* h3b  = (bf16_t*)take((size_t)N * h3 * 2);
    bf16_t* bufA = (bf16_t*)take((size_t)N * maxF * 2);
    float*  hw   = (float*)take((size_t)N * maxF * 4);
    float*  agg  = (float*)take((size_t)N * maxF * 4);
    if (off > ws_size) return;  // workspace too small: bail (nothing launched)

    const int T = 256;
    auto blks = [](long long n, int t) { return (unsigned)((n + t - 1) / t); };

    // 1) symmetric normalization coefficients
    k_init_deg<<<blks(N, T), T, 0, stream>>>(dinv, N);
    k_count_deg<<<blks(E, T), T, 0, stream>>>(ei + E, dinv, E);
    k_finalize_dinv<<<blks(N, T), T, 0, stream>>>(dinv, N);

    // 2) weights -> bf16, N-major
    k_transpose_w<<<blks((long long)d  * h1, T), T, 0, stream>>>(W1, W1t, d,  h1);
    k_transpose_w<<<blks((long long)h1 * h2, T), T, 0, stream>>>(W2, W2t, h1, h2);
    k_transpose_w<<<blks((long long)h2 * h3, T), T, 0, stream>>>(Wl, Wlt, h2, h3);

    // 3) initial node features
    k_build_h0<<<blks((long long)N * d, T), T, 0, stream>>>(x, emb, bufA, num_users, d, N * d);

    // ---- GCN layer 1: 16384x128 @ 128x256 ----
    {
        int tilesM = N / 16, tilesN = h1 / 16;
        gemm_bf16_wmma<<<blks((long long)tilesM * tilesN * 32, T), T, 0, stream>>>(
            bufA, W1t, hw, tilesM, tilesN, d, h1);
        int tot = N * h1;
        k_agg_self<<<blks(tot, T), T, 0, stream>>>(hw, dinv, agg, h1, tot);
        k_agg_edges<<<blks((long long)E * (h1 / 4), T), T, 0, stream>>>(
            ei, ei + E, hw, dinv, agg, E, h1 / 4, h1);
        k_bias_relu_bf16<<<blks(tot, T), T, 0, stream>>>(agg, b1, bufA, h1 - 1, tot);
    }

    // ---- GCN layer 2: 16384x256 @ 256x128 ----
    {
        int tilesM = N / 16, tilesN = h2 / 16;
        gemm_bf16_wmma<<<blks((long long)tilesM * tilesN * 32, T), T, 0, stream>>>(
            bufA, W2t, hw, tilesM, tilesN, h1, h2);
        int tot = N * h2;
        k_agg_self<<<blks(tot, T), T, 0, stream>>>(hw, dinv, agg, h2, tot);
        k_agg_edges<<<blks((long long)E * (h2 / 4), T), T, 0, stream>>>(
            ei, ei + E, hw, dinv, agg, E, h2 / 4, h2);
        k_bias_relu_bf16<<<blks(tot, T), T, 0, stream>>>(agg, b2, bufA, h2 - 1, tot);
    }

    // ---- final linear: 16384x128 @ 128x64, relu -> bf16 ----
    {
        int tilesM = N / 16, tilesN = h3 / 16;
        gemm_bf16_wmma<<<blks((long long)tilesM * tilesN * 32, T), T, 0, stream>>>(
            bufA, Wlt, hw, tilesM, tilesN, h2, h3);
        int tot = N * h3;
        k_bias_relu_bf16<<<blks(tot, T), T, 0, stream>>>(hw, bl, h3b, h3 - 1, tot);
    }

    // ---- scores: clip(users @ items^T, 1, 5) -> d_out (8192x8192 f32) ----
    {
        const bf16_t* U = h3b;
        const bf16_t* I = h3b + (size_t)num_users * h3;
        int tilesM = num_users / 16, tilesN4 = num_items / 64;
        gemm_bf16_wmma_clip4<<<blks((long long)tilesM * tilesN4 * 32, T), T, 0, stream>>>(
            U, I, out, tilesM, tilesN4, h3, num_items);
    }
}
